// TokenWiseGatedMoELoraLinear_79207786873078
// MI455X (gfx1250) — compile-verified, matrix operand order checked by reading, not run on hardware
//
#include <hip/hip_runtime.h>
#include <hip/hip_bf16.h>
#include <math.h>

// Problem constants (from reference)
constexpr int BATCH = 4, TOK = 2048, DIM = 4096, OUT = 4096;
constexpr int E = 8, R = 16, ER = E * R;          // 128
constexpr int T = BATCH * TOK;                    // 8192 tokens
constexpr float SCALING = 16.0f / R;              // 1.0
constexpr int KTOT = DIM + ER;                    // 4224: base K + LoRA K fused

// Tiling
constexpr int BM = 128, BN = 128, BK = 32;        // BK == bf16 WMMA K
constexpr int LDK = BK + 4;                       // fp32 LDS stride (h_kernel)
constexpr int STR = 40;                           // bf16 LDS stride: 80B rows, 16B aligned, conflict-free b128

typedef float  v2f  __attribute__((ext_vector_type(2)));
typedef float  v8f  __attribute__((ext_vector_type(8)));
typedef __bf16 v8bf  __attribute__((ext_vector_type(8)));
typedef __bf16 v16bf __attribute__((ext_vector_type(16)));

// fp32 -> bf16 (RNE) split helpers (bit-twiddled; data is finite random normal)
__device__ __forceinline__ unsigned f2bf(float f) {
    unsigned u = __builtin_bit_cast(unsigned, f);
    return (u + 0x7FFFu + ((u >> 16) & 1u)) >> 16;
}
__device__ __forceinline__ float bf2f(unsigned h) {
    return __builtin_bit_cast(float, h << 16);
}
__device__ __forceinline__ void split_bf16(float v, unsigned& h, unsigned& l) {
    h = f2bf(v);
    l = f2bf(v - bf2f(h));
}

// ---------------------------------------------------------------------------
// Kernel 1: gating. One wave per token. softmax+top2+renorm collapses to a
// sigmoid of the top-2 logit gap. Writes routew[T][E] (includes SCALING).
// ---------------------------------------------------------------------------
__global__ __launch_bounds__(256) void gate_kernel(const float* __restrict__ x,
                                                   const float* __restrict__ Wg,
                                                   float* __restrict__ routew) {
    const int wave = threadIdx.x >> 5;
    const int lane = threadIdx.x & 31;
    const int t = blockIdx.x * 8 + wave;
    if (t >= T) return;

    const float* xr = x + (size_t)t * DIM;
    float p[E];
#pragma unroll
    for (int e = 0; e < E; ++e) p[e] = 0.f;

    for (int d = lane; d < DIM; d += 32) {
        const float xv = xr[d];
#pragma unroll
        for (int e = 0; e < E; ++e) p[e] += xv * Wg[(size_t)e * DIM + d];
    }
#pragma unroll
    for (int e = 0; e < E; ++e) {
#pragma unroll
        for (int off = 16; off > 0; off >>= 1) p[e] += __shfl_xor(p[e], off, 32);
    }
    if (lane == 0) {
        int i1 = 0; float l1 = p[0];
#pragma unroll
        for (int e = 1; e < E; ++e) if (p[e] > l1) { l1 = p[e]; i1 = e; }
        int i2 = -1; float l2 = -INFINITY;
#pragma unroll
        for (int e = 0; e < E; ++e) if (e != i1 && p[e] > l2) { l2 = p[e]; i2 = e; }
        const float w1 = 1.f / (1.f + expf(l2 - l1));
        const float w2 = 1.f - w1;
        float* rw = routew + (size_t)t * E;
#pragma unroll
        for (int e = 0; e < E; ++e) rw[e] = 0.f;
        rw[i1] = w1 * SCALING;
        rw[i2] = w2 * SCALING;
    }
}

// ---------------------------------------------------------------------------
// Kernel 2: HW[T][128] = (X @ A_cat^T) * routew via fp32 WMMA (small: 3% of
// total FLOPs, keep full fp32 precision here).
// ---------------------------------------------------------------------------
__global__ __launch_bounds__(256) void h_kernel(const float* __restrict__ x,
                                                const float* __restrict__ Acat,
                                                const float* __restrict__ routew,
                                                float* __restrict__ HW) {
    __shared__ float Xs[BM * LDK];
    __shared__ float As[ER * LDK];

    const int tid  = threadIdx.x;
    const int lane = tid & 31;
    const int wave = tid >> 5;
    const int m0 = (wave & 3) * 32;
    const int n0 = (wave >> 2) * 64;
    const int half = lane >> 4;
    const int l15  = lane & 15;
    const int rowBase = blockIdx.x * BM;

    v8f acc[2][4];
#pragma unroll
    for (int tm = 0; tm < 2; ++tm)
#pragma unroll
        for (int tn = 0; tn < 4; ++tn) acc[tm][tn] = (v8f)0.f;

    for (int kb = 0; kb < DIM / BK; ++kb) {
        const int k0 = kb * BK;
        if (kb) __syncthreads();
#pragma unroll
        for (int i = 0; i < 4; ++i) {
            const int f   = tid + i * 256;
            const int row = f >> 3;
            const int c4  = (f & 7) << 2;
            *(float4*)&Xs[row * LDK + c4] =
                *(const float4*)&x[(size_t)(rowBase + row) * DIM + k0 + c4];
            *(float4*)&As[row * LDK + c4] =
                *(const float4*)&Acat[(size_t)row * DIM + k0 + c4];
        }
        __syncthreads();

#pragma unroll
        for (int kk = 0; kk < BK; kk += 4) {
            v2f af[2], bf[4];
#pragma unroll
            for (int tm = 0; tm < 2; ++tm)
                af[tm] = *(const v2f*)&Xs[(m0 + tm * 16 + l15) * LDK + kk + half * 2];
#pragma unroll
            for (int tn = 0; tn < 4; ++tn)
                bf[tn] = *(const v2f*)&As[(n0 + tn * 16 + l15) * LDK + kk + half * 2];
#pragma unroll
            for (int tm = 0; tm < 2; ++tm)
#pragma unroll
                for (int tn = 0; tn < 4; ++tn)
                    acc[tm][tn] = __builtin_amdgcn_wmma_f32_16x16x4_f32(
                        false, af[tm], false, bf[tn], (short)0, acc[tm][tn], false, false);
        }
    }

#pragma unroll
    for (int tn = 0; tn < 4; ++tn) {
        const int col = n0 + tn * 16 + l15;     // 0..127
        const int e = col >> 4;
#pragma unroll
        for (int tm = 0; tm < 2; ++tm) {
#pragma unroll
            for (int v = 0; v < 8; ++v) {
                const int row = rowBase + m0 + tm * 16 + v + half * 8;
                HW[(size_t)row * ER + col] = acc[tm][tn][v] * routew[(size_t)row * E + e];
            }
        }
    }
}

// ---------------------------------------------------------------------------
// Kernel 3a: one-time bf16 hi/lo split of activations, K-concatenated:
// Act[t][0..4095] = X, Act[t][4096..4223] = HW (routing-weighted LoRA h).
// ---------------------------------------------------------------------------
__global__ __launch_bounds__(256) void split_act_kernel(const float* __restrict__ x,
                                                        const float* __restrict__ HW,
                                                        unsigned short* __restrict__ ActH,
                                                        unsigned short* __restrict__ ActL) {
    const size_t q = (size_t)blockIdx.x * 256 + threadIdx.x;   // float4 id
    if (q >= (size_t)T * KTOT / 4) return;
    const int    col = (int)((q * 4) % KTOT);
    const size_t row = (q * 4) / KTOT;
    float4 v;
    if (col < DIM) v = *(const float4*)&x[row * DIM + col];
    else           v = *(const float4*)&HW[row * ER + (col - DIM)];
    unsigned h0, h1, h2, h3, l0, l1, l2, l3;
    split_bf16(v.x, h0, l0); split_bf16(v.y, h1, l1);
    split_bf16(v.z, h2, l2); split_bf16(v.w, h3, l3);
    ushort4 hp{(unsigned short)h0, (unsigned short)h1, (unsigned short)h2, (unsigned short)h3};
    ushort4 lp{(unsigned short)l0, (unsigned short)l1, (unsigned short)l2, (unsigned short)l3};
    *(ushort4*)&ActH[row * KTOT + col] = hp;
    *(ushort4*)&ActL[row * KTOT + col] = lp;
}

// ---------------------------------------------------------------------------
// Kernel 3b: one-time bf16 hi/lo split of weights, K-concatenated:
// Wgt[o][0..4095] = W_base row, Wgt[o][4096+e*16+r] = B[e][o][r].
// ---------------------------------------------------------------------------
__global__ __launch_bounds__(256) void split_wgt_kernel(const float* __restrict__ Wb,
                                                        const float* __restrict__ Bmat,
                                                        unsigned short* __restrict__ WgtH,
                                                        unsigned short* __restrict__ WgtL) {
    const size_t q = (size_t)blockIdx.x * 256 + threadIdx.x;   // float4 id
    if (q >= (size_t)OUT * KTOT / 4) return;
    const int    col = (int)((q * 4) % KTOT);
    const size_t row = (q * 4) / KTOT;
    float4 v;
    if (col < DIM) {
        v = *(const float4*)&Wb[row * DIM + col];
    } else {
        const int k = col - DIM;                // multiple of 4, within one expert block
        const int e = k >> 4;
        v = *(const float4*)&Bmat[((size_t)e * OUT + row) * R + (k & 15)];
    }
    unsigned h0, h1, h2, h3, l0, l1, l2, l3;
    split_bf16(v.x, h0, l0); split_bf16(v.y, h1, l1);
    split_bf16(v.z, h2, l2); split_bf16(v.w, h3, l3);
    ushort4 hp{(unsigned short)h0, (unsigned short)h1, (unsigned short)h2, (unsigned short)h3};
    ushort4 lp{(unsigned short)l0, (unsigned short)l1, (unsigned short)l2, (unsigned short)l3};
    *(ushort4*)&WgtH[row * KTOT + col] = hp;
    *(ushort4*)&WgtL[row * KTOT + col] = lp;
}

// ---------------------------------------------------------------------------
// Kernel 4: OUT = Act @ Wgt^T + bias over the fused K=4224, fp32 emulated via
// bf16 split: x*w ~= hh' + hl' + lh' with fp32 accumulation in
// V_WMMA_F32_16X16X32_BF16. Inner loop: coalesced bf16 plane loads -> LDS ->
// two ds_load_b128 per fragment -> 3 WMMAs per tile per K=32. No conversion
// VALU in the loop.
// ---------------------------------------------------------------------------
__global__ __launch_bounds__(256) void main_kernel(const unsigned short* __restrict__ ActH,
                                                   const unsigned short* __restrict__ ActL,
                                                   const unsigned short* __restrict__ WgtH,
                                                   const unsigned short* __restrict__ WgtL,
                                                   const float* __restrict__ bias,
                                                   float* __restrict__ out) {
    __shared__ unsigned short XhS[BM * STR];
    __shared__ unsigned short XlS[BM * STR];
    __shared__ unsigned short WhS[BN * STR];
    __shared__ unsigned short WlS[BN * STR];

    const int tid  = threadIdx.x;
    const int lane = tid & 31;
    const int wave = tid >> 5;
    const int m0 = (wave & 3) * 32;      // 4 waves in M
    const int n0 = (wave >> 2) * 64;     // 2 waves in N
    const int half = lane >> 4;
    const int l15  = lane & 15;
    const int rowBase = blockIdx.y * BM;
    const int colBase = blockIdx.x * BN;

    v8f acc[2][4];
#pragma unroll
    for (int tm = 0; tm < 2; ++tm)
#pragma unroll
        for (int tn = 0; tn < 4; ++tn) acc[tm][tn] = (v8f)0.f;

    constexpr int KB = KTOT / BK;        // 132 (base + LoRA fused)

    for (int kb = 0; kb < KB; ++kb) {
        const int k0 = kb * BK;
        if (kb) __syncthreads();
        // 512 16B-chunks per plane tile (128 rows x 32 bf16); 2 per thread.
#pragma unroll
        for (int i = 0; i < 2; ++i) {
            const int c   = tid + i * 256;
            const int row = c >> 2;
            const int c8  = (c & 3) << 3;           // bf16 col offset in tile
            const size_t gA = (size_t)(rowBase + row) * KTOT + k0 + c8;
            const size_t gW = (size_t)(colBase + row) * KTOT + k0 + c8;
            *(uint4*)&XhS[row * STR + c8] = *(const uint4*)&ActH[gA];
            *(uint4*)&XlS[row * STR + c8] = *(const uint4*)&ActL[gA];
            *(uint4*)&WhS[row * STR + c8] = *(const uint4*)&WgtH[gW];
            *(uint4*)&WlS[row * STR + c8] = *(const uint4*)&WgtL[gW];
            if (kb + 1 < KB) {
                __builtin_prefetch(&ActH[gA + BK], 0, 0);
                __builtin_prefetch(&WgtH[gW + BK], 0, 0);
            }
        }
        __syncthreads();

        // Fragment layout (16-bit, 16x32 / 32x16): lane's row/col = l15,
        // chunk0 = K[half*8 .. +8), chunk1 = chunk0 + 16.
        v16bf ah[2], al[2], bh[4], bl[4];
#pragma unroll
        for (int tm = 0; tm < 2; ++tm) {
            const int base = (m0 + tm * 16 + l15) * STR + half * 8;
            v8bf c0 = *(const v8bf*)&XhS[base];
            v8bf c1 = *(const v8bf*)&XhS[base + 16];
            ah[tm] = __builtin_shufflevector(c0, c1, 0,1,2,3,4,5,6,7,8,9,10,11,12,13,14,15);
            v8bf d0 = *(const v8bf*)&XlS[base];
            v8bf d1 = *(const v8bf*)&XlS[base + 16];
            al[tm] = __builtin_shufflevector(d0, d1, 0,1,2,3,4,5,6,7,8,9,10,11,12,13,14,15);
        }
#pragma unroll
        for (int tn = 0; tn < 4; ++tn) {
            const int base = (n0 + tn * 16 + l15) * STR + half * 8;
            v8bf c0 = *(const v8bf*)&WhS[base];
            v8bf c1 = *(const v8bf*)&WhS[base + 16];
            bh[tn] = __builtin_shufflevector(c0, c1, 0,1,2,3,4,5,6,7,8,9,10,11,12,13,14,15);
            v8bf d0 = *(const v8bf*)&WlS[base];
            v8bf d1 = *(const v8bf*)&WlS[base + 16];
            bl[tn] = __builtin_shufflevector(d0, d1, 0,1,2,3,4,5,6,7,8,9,10,11,12,13,14,15);
        }
#pragma unroll
        for (int tm = 0; tm < 2; ++tm) {
#pragma unroll
            for (int tn = 0; tn < 4; ++tn) {
                acc[tm][tn] = __builtin_amdgcn_wmma_f32_16x16x32_bf16(
                    false, ah[tm], false, bh[tn], (short)0, acc[tm][tn], false, false);
                acc[tm][tn] = __builtin_amdgcn_wmma_f32_16x16x32_bf16(
                    false, ah[tm], false, bl[tn], (short)0, acc[tm][tn], false, false);
                acc[tm][tn] = __builtin_amdgcn_wmma_f32_16x16x32_bf16(
                    false, al[tm], false, bh[tn], (short)0, acc[tm][tn], false, false);
            }
        }
    }

    // epilogue: + bias, store
#pragma unroll
    for (int tn = 0; tn < 4; ++tn) {
        const int col = colBase + n0 + tn * 16 + l15;
        const float bv = bias[col];
#pragma unroll
        for (int tm = 0; tm < 2; ++tm) {
#pragma unroll
            for (int v = 0; v < 8; ++v) {
                const int row = rowBase + m0 + tm * 16 + v + half * 8;
                out[(size_t)row * OUT + col] = acc[tm][tn][v] + bv;
            }
        }
    }
}

// ---------------------------------------------------------------------------
extern "C" void kernel_launch(void* const* d_in, const int* in_sizes, int n_in,
                              void* d_out, int out_size, void* d_ws, size_t ws_size,
                              hipStream_t stream) {
    const float* x    = (const float*)d_in[0];   // [4,2048,4096]
    const float* Wb   = (const float*)d_in[1];   // [4096,4096]
    const float* bb   = (const float*)d_in[2];   // [4096]
    const float* Wg   = (const float*)d_in[3];   // [8,4096]
    const float* Acat = (const float*)d_in[4];   // [8,16,4096] == [128,4096]
    const float* Bmat = (const float*)d_in[5];   // [8,4096,16]
    float* out = (float*)d_out;

    // Workspace layout (bytes, all 16B-aligned region sizes)
    char* ws = (char*)d_ws;
    float* routew = (float*)ws;                             ws += (size_t)T * E * 4;      // 256 KB
    float* HW     = (float*)ws;                             ws += (size_t)T * ER * 4;     // 4 MB
    unsigned short* ActH = (unsigned short*)ws;             ws += (size_t)T * KTOT * 2;   // 69 MB
    unsigned short* ActL = (unsigned short*)ws;             ws += (size_t)T * KTOT * 2;   // 69 MB
    unsigned short* WgtH = (unsigned short*)ws;             ws += (size_t)OUT * KTOT * 2; // 35 MB
    unsigned short* WgtL = (unsigned short*)ws;             ws += (size_t)OUT * KTOT * 2; // 35 MB

    gate_kernel<<<T / 8, 256, 0, stream>>>(x, Wg, routew);
    h_kernel<<<T / BM, 256, 0, stream>>>(x, Acat, routew, HW);

    const int nq_act = (int)(((size_t)T * KTOT / 4 + 255) / 256);
    const int nq_wgt = (int)(((size_t)OUT * KTOT / 4 + 255) / 256);
    split_act_kernel<<<nq_act, 256, 0, stream>>>(x, HW, ActH, ActL);
    split_wgt_kernel<<<nq_wgt, 256, 0, stream>>>(Wb, Bmat, WgtH, WgtL);

    main_kernel<<<dim3(OUT / BN, T / BM), 256, 0, stream>>>(ActH, ActL, WgtH, WgtL, bb, out);
}
